// OctonionSSM_58918361366616
// MI455X (gfx1250) — matrix operand order repaired and checked by reading, not compiled, not run on hardware
//
#include <hip/hip_runtime.h>
#include <hip/hip_bf16.h>

// ---------------------------------------------------------------------------
// Octonion SSM for MI455X (gfx1250, wave32, WMMA).
//   B=8, T=1024, D=1024, S=2048.
//   Pipeline: fold weights -> bf16 WMMA GEMMs (delta w/ sigmoid, Bx) ->
//             persistent grid-synced scan (h@Af^T via WMMA, Af slice in LDS) ->
//             readout GEMM y = hs @ Cf^T (f32 out).
//   Tile staging uses GLOBAL_LOAD_ASYNC_TO_LDS_B128 (ASYNCcnt) double-buffered.
// ---------------------------------------------------------------------------

typedef __bf16 bf16;
typedef __attribute__((ext_vector_type(8)))  __bf16 v8bf;
typedef __attribute__((ext_vector_type(16))) __bf16 v16bf;
typedef __attribute__((ext_vector_type(8)))  float  v8f;

#define SHUF16(lo, hi) __builtin_shufflevector(lo, hi, 0,1,2,3,4,5,6,7,8,9,10,11,12,13,14,15)

// LDS pitches (bf16 elems). Byte pitches 80 / 144: 16B-aligned rows for
// async b128 LDS writes, and 20-/36-bank strides (distinct bank phases).
#define LDA_P 40
#define LDB_P 72

// Async global->LDS 16-byte copy (CDNA5 GLOBAL_LOAD_ASYNC_TO_LDS_B128).
// LDS address = low 32 bits of the generic pointer (LDS aperture mapping).
__device__ __forceinline__ void async_copy16(const bf16* g, bf16* l) {
  unsigned lo = (unsigned)(size_t)(void*)l;
  asm volatile("global_load_async_to_lds_b128 %0, %1, off"
               :: "v"(lo), "v"(g) : "memory");
}
__device__ __forceinline__ void async_wait0() {
  asm volatile("s_wait_asynccnt 0x0" ::: "memory");
}

// Octonion multiplication tables: e_i e_j = sgn * e_idx  (idx[i*8+j], sgn[i*8+j])
__device__ static const int OCT_IDX[64] = {
  0,1,2,3,4,5,6,7,
  1,0,4,7,2,6,5,3,
  2,4,0,5,1,3,7,6,
  3,7,5,0,6,2,4,1,
  4,2,1,6,0,7,3,5,
  5,6,3,2,7,0,1,4,
  6,5,7,4,3,1,0,2,
  7,3,6,1,5,4,2,0 };
__device__ static const float OCT_SGN[64] = {
  1, 1, 1, 1, 1, 1, 1, 1,
  1,-1, 1, 1,-1, 1,-1,-1,
  1,-1,-1, 1, 1,-1, 1,-1,
  1,-1,-1,-1, 1, 1,-1, 1,
  1, 1,-1,-1,-1, 1, 1,-1,
  1,-1, 1,-1,-1,-1, 1, 1,
  1, 1,-1, 1,-1,-1,-1, 1,
  1, 1, 1,-1, 1,-1,-1,-1 };

// -------------------------- small prep kernels -----------------------------

__global__ void k_convert_x(const float* __restrict__ x, bf16* __restrict__ xb) {
  int t0 = (blockIdx.x * 256 + threadIdx.x) * 4;
  float4 v = *(const float4*)(x + t0);
  xb[t0 + 0] = (bf16)v.x; xb[t0 + 1] = (bf16)v.y;
  xb[t0 + 2] = (bf16)v.z; xb[t0 + 3] = (bf16)v.w;
}

// Wdelta [2048,1024] -> WdeltaT [1024,2048] bf16
__global__ void k_transpose_wdelta(const float* __restrict__ Wd, bf16* __restrict__ out) {
  int e = blockIdx.x * 256 + threadIdx.x;   // 2^21 elements
  int d = e >> 11, s = e & 2047;
  out[e] = (bf16)Wd[s * 1024 + d];
}

// Fold octonion structure, producing the TRANSPOSED dense matrix directly:
//   out[(j*ND+n) * (8*OD) + k*OD+o] = sgn(j,k) * W[i(j,k), o, n]
__global__ void k_oct_fold(const float* __restrict__ W, bf16* __restrict__ out,
                           int OD, int ND) {
  const int COLS = 8 * OD;
  const long total = (long)8 * ND * COLS;
  for (long e = (long)blockIdx.x * blockDim.x + threadIdx.x; e < total;
       e += (long)gridDim.x * blockDim.x) {
    int row = (int)(e / COLS), col = (int)(e % COLS);
    int j = row / ND, n = row % ND;
    int k = col / OD, o = col % OD;
    int ii = 0; float s = 0.f;
    #pragma unroll
    for (int i = 0; i < 8; ++i)
      if (OCT_IDX[i * 8 + j] == k) { ii = i; s = OCT_SGN[i * 8 + j]; }
    out[e] = (bf16)(s * W[((long)ii * OD + o) * ND + n]);
  }
}

__global__ void k_scan_init(const float* __restrict__ hprev, float* __restrict__ hf,
                            bf16* __restrict__ hb, unsigned* __restrict__ sync) {
  int e = blockIdx.x * 256 + threadIdx.x;   // 32768 threads
  if (e < 16384) { float v = hprev[e]; hf[e] = v; hb[e] = (bf16)v; }
  else           { hb[e] = (bf16)0.f; }      // pad rows 8..15 of h
  if (e == 0) sync[0] = 0u;
}

// ------------------------------ WMMA GEMM ----------------------------------
// C[M,N] = act(A[M,K] * Bm[K,N]).  Tiles: 128x64x32, 8 waves (4x2), each wave
// 2x2 accumulators of v_wmma_f32_16x16x32_bf16. Double-buffered async LDS.
// MODE 0: bf16 store      MODE 1: sigmoid(.+bias) bf16 store
// MODE 2: f32 store with row remap r=t*8+b -> (b*Tdim+t)   (readout GEMM)

template <int MODE>
__launch_bounds__(256)
__global__ void k_gemm(const bf16* __restrict__ A, const bf16* __restrict__ Bm,
                       const float* __restrict__ bias, void* __restrict__ Cout,
                       int M, int N, int K, int Tdim) {
  __shared__ bf16 As[2][128 * LDA_P];
  __shared__ bf16 Bs[2][32 * LDB_P];

  const int tid = threadIdx.x, lane = tid & 31, wave = tid >> 5;
  const int wm = wave & 3, wn = wave >> 2;
  const long m0 = (long)blockIdx.x * 128;
  const long n0 = (long)blockIdx.y * 64;

  v8f acc[2][2] = {};

  const int mrow = lane & 15;
  const int kh   = (lane >> 4) * 8;

  const int ar0 = tid >> 2, ap = (tid & 3) * 8;   // A tile rows 0..63
  const int ar1 = ar0 + 64;                       // A tile rows 64..127
  const int br  = tid >> 3, bp = (tid & 7) * 8;   // B tile 32x64

  // prologue: async-stage chunk 0
  {
    async_copy16(A + (m0 + ar0) * K + ap,       &As[0][ar0 * LDA_P + ap]);
    async_copy16(A + (m0 + ar1) * K + ap,       &As[0][ar1 * LDA_P + ap]);
    async_copy16(Bm + (long)br * N + n0 + bp,   &Bs[0][br * LDB_P + bp]);
    async_wait0();
    __syncthreads();
  }

  int buf = 0;
  for (int kc = 0; kc < K; kc += 32) {
    if (kc + 32 < K) {   // issue next tile while computing this one
      int nb = buf ^ 1, kn = kc + 32;
      async_copy16(A + (m0 + ar0) * K + kn + ap,      &As[nb][ar0 * LDA_P + ap]);
      async_copy16(A + (m0 + ar1) * K + kn + ap,      &As[nb][ar1 * LDA_P + ap]);
      async_copy16(Bm + (long)(kn + br) * N + n0 + bp,&Bs[nb][br * LDB_P + bp]);
    }

    v16bf af[2], bfr[2];
    #pragma unroll
    for (int mt = 0; mt < 2; ++mt) {
      const bf16* p = &As[buf][(wm * 32 + mt * 16 + mrow) * LDA_P + kh];
      v8bf lo = *(const v8bf*)p, hi = *(const v8bf*)(p + 16);
      af[mt] = SHUF16(lo, hi);
    }
    #pragma unroll
    for (int nt = 0; nt < 2; ++nt) {
      const bf16* p = &Bs[buf][kh * LDB_P + wn * 32 + nt * 16 + mrow];
      #pragma unroll
      for (int s = 0; s < 8; ++s) {
        bfr[nt][s] = p[s * LDB_P];
        bfr[nt][s + 8] = p[(s + 16) * LDB_P];
      }
    }
    #pragma unroll
    for (int mt = 0; mt < 2; ++mt)
      #pragma unroll
      for (int nt = 0; nt < 2; ++nt)
        acc[mt][nt] = __builtin_amdgcn_wmma_f32_16x16x32_bf16(
            false, af[mt], false, bfr[nt], (short)0, acc[mt][nt], false, false);

    async_wait0();      // next tile resident
    __syncthreads();    // all waves done reading buf + all asyncs visible
    buf ^= 1;
  }

  #pragma unroll
  for (int mt = 0; mt < 2; ++mt)
    #pragma unroll
    for (int nt = 0; nt < 2; ++nt) {
      long row0 = m0 + wm * 32 + mt * 16 + (lane >> 4) * 8;
      long col  = n0 + wn * 32 + nt * 16 + (lane & 15);
      #pragma unroll
      for (int v = 0; v < 8; ++v) {
        long row = row0 + v;
        float val = acc[mt][nt][v];
        if (MODE == 0) {
          ((bf16*)Cout)[row * N + col] = (bf16)val;
        } else if (MODE == 1) {
          float xx = val + bias[col];
          float e  = __expf(-xx);
          ((bf16*)Cout)[row * N + col] = (bf16)__builtin_amdgcn_rcpf(1.f + e);
        } else {
          long orow = (row & 7) * Tdim + (row >> 3);
          ((float*)Cout)[orow * N + col] = val;
        }
      }
    }
}

// ------------------------------- scan --------------------------------------
// 64 persistent WGs x 256 threads. WG b owns state columns [b*32, b*32+32).
// Af^T column slice (2048 x 32 bf16, pitch 40 = 160KB) lives in LDS.
// Per step: 8 waves split K (2048) -> 8 chunks each -> 16 WMMAs/wave,
// LDS cross-wave reduction, gated f32 update, device-wide barrier.

#define SCAN_ASL_BYTES (2048 * LDA_P * 2)
#define SCAN_SMEM (SCAN_ASL_BYTES + 8 * 2 * 256 * 4)

__launch_bounds__(256)
__global__ void k_scan(const bf16* __restrict__ AfT, const bf16* __restrict__ dlt,
                       const bf16* __restrict__ Bx, float* __restrict__ hf,
                       bf16* __restrict__ hb, bf16* __restrict__ hs,
                       float* __restrict__ hlast, unsigned* __restrict__ sync,
                       int T) {
  extern __shared__ char smem[];
  bf16*  Asl = (bf16*)smem;                      // [2048][LDA_P]
  float* red = (float*)(smem + SCAN_ASL_BYTES);  // 8 waves * 2 tiles * 256 f32

  const int tid = threadIdx.x, lane = tid & 31, wave = tid >> 5;
  const int c0 = blockIdx.x * 32;

  // Stage Af^T slice once via async copies (resident for all 1024 steps)
  for (int it = 0; it < 32; ++it) {
    int idx = tid + it * 256;
    int r = idx >> 2, p = (idx & 3) * 8;
    async_copy16(AfT + (long)r * 2048 + c0 + p, Asl + r * LDA_P + p);
  }
  async_wait0();
  __syncthreads();

  const int mrow = lane & 15, kh = (lane >> 4) * 8;

  for (int t = 0; t < T; ++t) {
    v8f a0 = {}, a1 = {};
    #pragma unroll
    for (int q = 0; q < 8; ++q) {
      int kc = (wave * 8 + q) * 32;
      const bf16* ap = hb + mrow * 2048 + kc + kh;          // h (bf16, 16 rows)
      v8bf lo = *(const v8bf*)ap, hi = *(const v8bf*)(ap + 16);
      v16bf af = SHUF16(lo, hi);
      const bf16* bp = Asl + (kc + kh) * LDA_P + mrow;      // Af^T column mrow from LDS
      v16bf bf0, bf1;
      #pragma unroll
      for (int s = 0; s < 8; ++s) {
        bf0[s] = bp[s * LDA_P];            bf0[s + 8] = bp[(s + 16) * LDA_P];
        bf1[s] = bp[s * LDA_P + 16];       bf1[s + 8] = bp[(s + 16) * LDA_P + 16];
      }
      a0 = __builtin_amdgcn_wmma_f32_16x16x32_bf16(false, af, false, bf0, (short)0, a0, false, false);
      a1 = __builtin_amdgcn_wmma_f32_16x16x32_bf16(false, af, false, bf1, (short)0, a1, false, false);
    }
    {
      float* r0 = red + (wave * 2 + 0) * 256;
      float* r1 = red + (wave * 2 + 1) * 256;
      int cc = lane & 15, rb = (lane >> 4) * 8;
      #pragma unroll
      for (int v = 0; v < 8; ++v) {
        r0[(rb + v) * 16 + cc] = a0[v];
        r1[(rb + v) * 16 + cc] = a1[v];
      }
    }
    __syncthreads();
    {
      int m = tid >> 5, c = tid & 31;         // m: batch 0..7, c: local column
      int nt = c >> 4, cc = c & 15;
      float sum = 0.f;
      #pragma unroll
      for (int w2 = 0; w2 < 8; ++w2) sum += red[(w2 * 2 + nt) * 256 + m * 16 + cc];
      int col = c0 + c;
      long rb = ((long)m * T + t) * 2048 + col;
      float d  = (float)dlt[rb];
      float bx = (float)Bx[rb];
      float hold = hf[m * 2048 + col];
      float hn = (1.f - d) * hold + d * (sum + bx);
      hf[m * 2048 + col] = hn;
      hb[m * 2048 + col] = (bf16)hn;
      hs[((long)t * 8 + m) * 2048 + col] = (bf16)hn;
    }
    // device-wide barrier (release/acquire -> global_wb / global_inv)
    __syncthreads();
    if (tid == 0) {
      __hip_atomic_fetch_add(sync, 1u, __ATOMIC_RELEASE, __HIP_MEMORY_SCOPE_AGENT);
      unsigned target = 64u * (unsigned)(t + 1);
      while (__hip_atomic_load(sync, __ATOMIC_ACQUIRE, __HIP_MEMORY_SCOPE_AGENT) < target)
        __builtin_amdgcn_s_sleep(2);
    }
    __syncthreads();
  }
  {
    int m = tid >> 5, c = tid & 31;
    hlast[m * 2048 + c0 + c] = hf[m * 2048 + c0 + c];
  }
}

// ---------------------------- workspace map --------------------------------
// bytes (all 256-aligned); total ~132.2 MB
#define WS_XBF   ((size_t)0)            // x bf16            16,777,216
#define WS_WDT   ((size_t)16777216)     // WdeltaT bf16       4,194,304
#define WS_BFT   ((size_t)20971520)     // Bf^T bf16          4,194,304
#define WS_CFT   ((size_t)25165824)     // Cf^T bf16          4,194,304
#define WS_AFT   ((size_t)29360128)     // Af^T bf16          8,388,608
#define WS_DLT   ((size_t)37748736)     // delta bf16        33,554,432
#define WS_BX    ((size_t)71303168)     // Bx bf16           33,554,432
#define WS_HS    ((size_t)104857600)    // hs bf16           33,554,432
#define WS_HF    ((size_t)138412032)    // h f32 [8,2048]        65,536
#define WS_HB    ((size_t)138477568)    // h bf16 [16,2048]      65,536
#define WS_SYNC  ((size_t)138543104)    // barrier counter          256

extern "C" void kernel_launch(void* const* d_in, const int* in_sizes, int n_in,
                              void* d_out, int out_size, void* d_ws, size_t ws_size,
                              hipStream_t stream) {
  const float* x     = (const float*)d_in[0];
  const float* hprev = (const float*)d_in[1];
  const float* WA    = (const float*)d_in[2];
  const float* WB    = (const float*)d_in[3];
  const float* WC    = (const float*)d_in[4];
  const float* Wd    = (const float*)d_in[5];
  const float* dbias = (const float*)d_in[6];

  float* y     = (float*)d_out;               // [8,1024,1024]
  float* hlast = y + (size_t)8 * 1024 * 1024; // [8,2048]

  char* ws = (char*)d_ws;
  bf16* xb  = (bf16*)(ws + WS_XBF);
  bf16* wdT = (bf16*)(ws + WS_WDT);
  bf16* bfT = (bf16*)(ws + WS_BFT);
  bf16* cfT = (bf16*)(ws + WS_CFT);
  bf16* afT = (bf16*)(ws + WS_AFT);
  bf16* dlt = (bf16*)(ws + WS_DLT);
  bf16* bx  = (bf16*)(ws + WS_BX);
  bf16* hs  = (bf16*)(ws + WS_HS);
  float* hf = (float*)(ws + WS_HF);
  bf16* hb  = (bf16*)(ws + WS_HB);
  unsigned* sync = (unsigned*)(ws + WS_SYNC);

  k_convert_x       <<<8192, 256, 0, stream>>>(x, xb);
  k_transpose_wdelta<<<8192, 256, 0, stream>>>(Wd, wdT);
  k_oct_fold        <<<4096, 256, 0, stream>>>(WA, afT, 256, 256);
  k_oct_fold        <<<2048, 256, 0, stream>>>(WB, bfT, 256, 128);
  k_oct_fold        <<<2048, 256, 0, stream>>>(WC, cfT, 128, 256);

  dim3 g1(64, 32);   // M=8192/128, N=2048/64
  k_gemm<1><<<g1, 256, 0, stream>>>(xb, wdT, dbias, dlt, 8192, 2048, 1024, 0);
  k_gemm<0><<<g1, 256, 0, stream>>>(xb, bfT, nullptr, bx, 8192, 2048, 1024, 0);

  k_scan_init<<<128, 256, 0, stream>>>(hprev, hf, hb, sync);

  hipFuncSetAttribute((const void*)k_scan,
                      hipFuncAttributeMaxDynamicSharedMemorySize, SCAN_SMEM);
  k_scan<<<64, 256, SCAN_SMEM, stream>>>(afT, dlt, bx, hf, hb, hs, hlast, sync, 1024);

  dim3 g2(64, 16);   // M=8192/128, N=1024/64
  k_gemm<2><<<g2, 256, 0, stream>>>(hs, cfT, nullptr, y, 8192, 1024, 2048, 1024);
}
